// StringLabelEncoder_73641509257609
// MI455X (gfx1250) — 8-bit matrix operand order repaired and checked by reading, compiled, not run on hardware
//
#include <hip/hip_runtime.h>

typedef __attribute__((ext_vector_type(8))) int v8i;

__device__ __forceinline__ int byte_sq_norm(int w) {
    int b0 = w & 0xFF;
    int b1 = (w >> 8) & 0xFF;
    int b2 = (w >> 16) & 0xFF;
    int b3 = (w >> 24) & 0xFF;
    return b0 * b0 + b1 * b1 + b2 * b2 + b3 * b3;
}

__global__ void cnorm_kernel(const int* __restrict__ table, int* __restrict__ cnorm,
                             int nClasses) {
    int c = blockIdx.x * blockDim.x + threadIdx.x;
    if (c < nClasses) {
        int s = 0;
#pragma unroll
        for (int w = 0; w < 8; ++w) s += byte_sq_norm(table[c * 8 + w]);
        cnorm[c] = s;
    }
}

__global__ void __launch_bounds__(128) match_kernel(const int* __restrict__ x,
                                                    const int* __restrict__ table,
                                                    const int* __restrict__ cnorm,
                                                    float* __restrict__ out,
                                                    int nRows, int nClasses) {
    __shared__ int xnorm_s[128];

    const int tid = threadIdx.x;
    const int blockRow0 = blockIdx.x * 128;

    {
        int rr = blockRow0 + tid;
        if (rr >= nRows) rr = nRows - 1;
        int s = 0;
#pragma unroll
        for (int w = 0; w < 8; ++w) s += byte_sq_norm(x[rr * 8 + w]);
        xnorm_s[tid] = s;
    }
    __syncthreads();

    const int lane = tid & 31;
    const int waveId = tid >> 5;
    const int half = lane >> 4;
    const int m = lane & 15;

    const int waveRow0 = blockRow0 + waveId * 32;

    int r0 = waveRow0 + m;
    if (r0 >= nRows) r0 = nRows - 1;
    int r1 = waveRow0 + 16 + m;
    if (r1 >= nRows) r1 = nRows - 1;

    const int* rp0 = x + (size_t)r0 * 8 + half * 2;
    const int* rp1 = x + (size_t)r1 * 8 + half * 2;
    v8i A0 = {}, A1 = {};
    A0[0] = rp0[0]; A0[1] = rp0[1]; A0[2] = rp0[4]; A0[3] = rp0[5];
    A1[0] = rp1[0]; A1[1] = rp1[1]; A1[2] = rp1[4]; A1[3] = rp1[5];

    int xn0[8], xn1[8];
#pragma unroll
    for (int r = 0; r < 8; ++r) {
        xn0[r] = xnorm_s[waveId * 32 + r + 8 * half];
        xn1[r] = xnorm_s[waveId * 32 + 16 + r + 8 * half];
    }

    const int nTiles = (nClasses + 15) / 16;
    for (int t = 0; t < nTiles; ++t) {
        int cls = t * 16 + m;
        int clsC = cls < nClasses ? cls : (nClasses - 1);
        const int* cp = table + (size_t)clsC * 8 + half * 2;
        v8i B = {};
        B[0] = cp[0]; B[1] = cp[1]; B[2] = cp[4]; B[3] = cp[5];
        int cn = cnorm[clsC];

        v8i Z = {};
        v8i D0 = __builtin_amdgcn_wmma_i32_16x16x64_iu8(false, A0, false, B, Z,
                                                        false, false);
        v8i D1 = __builtin_amdgcn_wmma_i32_16x16x64_iu8(false, A1, false, B, Z,
                                                        false, false);

        int sc0[8], sc1[8];
        bool any0 = false, any1 = false;
#pragma unroll
        for (int r = 0; r < 8; ++r) {
            sc0[r] = xn0[r] - 2 * D0[r] + cn;
            sc1[r] = xn1[r] - 2 * D1[r] + cn;
            any0 |= (sc0[r] == 0);
            any1 |= (sc1[r] == 0);
        }
        if (any0 && cls < nClasses) {
#pragma unroll
            for (int r = 0; r < 8; ++r) {
                int row = waveRow0 + r + 8 * half;
                if (sc0[r] == 0 && row < nRows) out[row] = (float)cls;
            }
        }
        if (any1 && cls < nClasses) {
#pragma unroll
            for (int r = 0; r < 8; ++r) {
                int row = waveRow0 + 16 + r + 8 * half;
                if (sc1[r] == 0 && row < nRows) out[row] = (float)cls;
            }
        }
    }
}

extern "C" void kernel_launch(void* const* d_in, const int* in_sizes, int n_in,
                              void* d_out, int out_size, void* d_ws, size_t ws_size,
                              hipStream_t stream) {
    const int* x = (const int*)d_in[0];
    const int* table = (const int*)d_in[1];
    float* out = (float*)d_out;
    int* cnorm = (int*)d_ws;

    const int nRows = in_sizes[0] / 8;
    const int nClasses = in_sizes[1] / 8;

    cnorm_kernel<<<(nClasses + 127) / 128, 128, 0, stream>>>(table, cnorm, nClasses);

    const int nBlocks = (nRows + 127) / 128;
    match_kernel<<<nBlocks, 128, 0, stream>>>(x, table, cnorm, out, nRows, nClasses);
}
